// NetGlobLocalGATFix_69569880261292
// MI455X (gfx1250) — matrix operand order repaired and checked by reading, not compiled, not run on hardware
//
#include <hip/hip_runtime.h>
#include <hip/hip_bf16.h>
#include <cstdint>

typedef __attribute__((ext_vector_type(16))) _Float16 v16h;
typedef __attribute__((ext_vector_type(8)))  _Float16 v8h;
typedef __attribute__((ext_vector_type(8)))  float    v8f;

#define NN 30000
#define EE 180000
#define ET (EE + NN)

__device__ __forceinline__ float selu_f(float v) {
    const float a = 1.6732632423543772f, sc = 1.0507009873554805f;
    return sc * (v > 0.f ? v : a * (__expf(v) - 1.f));
}
// order-preserving map float -> uint for atomicMax-based segment max
__device__ __forceinline__ unsigned enc_f(float f) {
    unsigned u = __float_as_uint(f);
    return (u & 0x80000000u) ? ~u : (u | 0x80000000u);
}
__device__ __forceinline__ float dec_f(unsigned u) {
    return (u & 0x80000000u) ? __uint_as_float(u ^ 0x80000000u) : __uint_as_float(~u);
}
__device__ __forceinline__ void edge_sd(const int* ei, int e, int& s, int& d) {
    if (e < EE) { s = ei[e]; d = ei[EE + e]; }
    else        { s = e - EE; d = s; }           // self loops appended
}

// ---------------- utility ----------------
__global__ void k_zero(float* p, int n) {
    int t = blockIdx.x * blockDim.x + threadIdx.x;
    if (t < n) p[t] = 0.f;
}

// ---------------- GFE: conv3x3 + SELU on 64x64 ----------------
__global__ void k_conv3x3_selu(const float* __restrict__ in, const float* __restrict__ w,
                               const float* __restrict__ b, float* __restrict__ out,
                               int Ci, int Co) {
    int t = blockIdx.x * blockDim.x + threadIdx.x;
    if (t >= Co * 4096) return;
    int co = t >> 12, p = t & 4095, y = p >> 6, x = p & 63;
    float s = b[co];
    for (int ci = 0; ci < Ci; ci++) {
        const float* ip = in + ci * 4096;
        const float* wp = w + (co * Ci + ci) * 9;
        for (int ky = -1; ky <= 1; ky++) {
            int yy = y + ky; if (yy < 0 || yy > 63) continue;
            for (int kx = -1; kx <= 1; kx++) {
                int xx = x + kx; if (xx < 0 || xx > 63) continue;
                s += ip[yy * 64 + xx] * wp[(ky + 1) * 3 + (kx + 1)];
            }
        }
    }
    out[t] = selu_f(s);
}

__global__ void k_avgpool16(const float* __restrict__ in, float* __restrict__ g) {
    __shared__ float sm[256];
    int c = blockIdx.x;
    float s = 0.f;
    for (int i = threadIdx.x; i < 4096; i += 256) s += in[c * 4096 + i];
    sm[threadIdx.x] = s;
    __syncthreads();
    for (int o = 128; o > 0; o >>= 1) {
        if ((int)threadIdx.x < o) sm[threadIdx.x] += sm[threadIdx.x + o];
        __syncthreads();
    }
    if (threadIdx.x == 0) g[c] = sm[0] * (1.f / 4096.f);
}

// ---------------- LFE edge feature build: [xi, xj, xj-xi] -> f16 padded to 32 ----
__global__ void k_build_m16(const float* __restrict__ x, const int* __restrict__ ei,
                            _Float16* __restrict__ m16) {
    int t = blockIdx.x * blockDim.x + threadIdx.x;
    if (t >= EE * 32) return;
    int e = t >> 5, j = t & 31;
    float v = 0.f;
    if (j < 30) {
        int s = ei[e], d = ei[EE + e];          // xi = x[dst], xj = x[src]
        if (j < 10)      v = x[d * 10 + j];
        else if (j < 20) v = x[s * 10 + (j - 10)];
        else             v = x[s * 10 + (j - 20)] - x[d * 10 + (j - 20)];
    }
    m16[t] = (_Float16)v;
}

// ---------------- weight repack into WMMA B-fragment order (LDS-friendly) ------
// frag f = kt*(Ncp/16)+nt, 512 halfs per frag stored as:
//   idx_in_frag = chunk*256 + lane*8 + e2   (chunk in {0,1}, e = chunk*8+e2)
// semantic element: col N = (lane&15), K = kt*32 + (lane>>4)*16 + e
// 16B per lane at 16B lane stride -> conflict-free ds_load_b128 later.
__global__ void k_repackB(const float* __restrict__ W, _Float16* __restrict__ Bp,
                          int K, int Nc, int Kp, int Ncp) {
    int t = blockIdx.x * blockDim.x + threadIdx.x;
    int total = (Kp >> 5) * (Ncp >> 4) * 512;
    if (t >= total) return;
    int f = t >> 9, r = t & 511;
    int chunk = r >> 8, lane = (r >> 3) & 31, e2 = r & 7;
    int e = chunk * 8 + e2;
    int nfr = Ncp >> 4;
    int kt = f / nfr, nt = f - kt * nfr;
    int k = kt * 32 + ((lane >> 4) << 4) + e;
    int n = nt * 16 + (lane & 15);
    float v = (k < K && n < Nc) ? W[k * Nc + n] : 0.f;
    Bp[t] = (_Float16)v;
}

// ---------------- f32 -> f16 with zero K padding ----------------
__global__ void k_f2h_pad(const float* __restrict__ src, _Float16* __restrict__ dst,
                          int rows, int K, int Kp) {
    int t = blockIdx.x * blockDim.x + threadIdx.x;
    if (t >= rows * Kp) return;
    int r = t / Kp, k = t - r * Kp;
    dst[t] = (_Float16)(k < K ? src[(size_t)r * K + k] : 0.f);
}

// ---------------- WMMA GEMM with async B staging in LDS -----------------------
// Block = 256 threads = 8 waves; block computes a 128(M) x 64(N) tile.
// B tile (4 n-frags x all K frags) is async-copied to LDS once, shared by waves.
__global__ void __launch_bounds__(256)
k_gemm_wmma(const _Float16* __restrict__ A, const _Float16* __restrict__ Bp,
            float* __restrict__ C, int M, int Kp, int Ncp) {
    __shared__ _Float16 smemB[16384];            // up to Kp=256: (Kp/32)*4*512 halfs
    int nt64 = Ncp >> 6;
    int mb = blockIdx.x / nt64;
    int ng = blockIdx.x - mb * nt64;
    int nfr = Ncp >> 4;
    int nK = Kp >> 5;
    int nChunks = (nK * 4) * 64;                 // 16B chunks in the B tile
    int tid = threadIdx.x;

    // async copy: global packed B -> LDS (layout preserved)
    for (int c = tid; c < nChunks; c += 256) {
        int fl = c >> 6, w = c & 63;             // frag-local, 16B chunk-in-frag
        int kt = fl >> 2, j = fl & 3;
        int gf = kt * nfr + (ng << 2) + j;
        const _Float16* gsrc = Bp + ((size_t)gf << 9) + (w << 3);
        unsigned loff = (unsigned)(uintptr_t)(&smemB[0]) + (unsigned)(fl * 1024 + w * 16);
        asm volatile("global_load_async_to_lds_b128 %0, %1, off"
                     :: "v"(loff), "v"(gsrc) : "memory");
    }
    asm volatile("s_wait_asynccnt 0x0" ::: "memory");
    __syncthreads();

    int wv = tid >> 5, lane = tid & 31;
    int mt = mb * 8 + wv;
    if (mt * 16 >= M) return;                    // wave-uniform exit (EXEC stays all-1)
    int m0 = mt * 16, n0 = ng << 6;
    int arow = m0 + (lane & 15);
    int koff = (lane >> 4) << 4;                 // lanes 16-31 take K+16 sub-rows
    const _Float16* abase = A + (size_t)arow * Kp + koff;
    v8f acc0 = {}, acc1 = {}, acc2 = {}, acc3 = {};
    for (int kt4 = 0; kt4 < nK; kt4++) {
        int kt = kt4 << 5;
        union AU { v16h v; v8h h[2]; } au, b0, b1, b2, b3;
        au.h[0] = *(const v8h*)(abase + kt);       // K = kt+koff .. +7
        au.h[1] = *(const v8h*)(abase + kt + 8);   // K = kt+8+koff .. +7
        if (kt + 32 < Kp) __builtin_prefetch(abase + kt + 32, 0, 3);
        const _Float16* fb = smemB + kt4 * 2048 + lane * 8;
        b0.h[0] = *(const v8h*)(fb);
        b0.h[1] = *(const v8h*)(fb + 256);
        b1.h[0] = *(const v8h*)(fb + 512);
        b1.h[1] = *(const v8h*)(fb + 768);
        b2.h[0] = *(const v8h*)(fb + 1024);
        b2.h[1] = *(const v8h*)(fb + 1280);
        b3.h[0] = *(const v8h*)(fb + 1536);
        b3.h[1] = *(const v8h*)(fb + 1792);
        acc0 = __builtin_amdgcn_wmma_f32_16x16x32_f16(false, au.v, false, b0.v, (short)0, acc0, false, false);
        acc1 = __builtin_amdgcn_wmma_f32_16x16x32_f16(false, au.v, false, b1.v, (short)0, acc1, false, false);
        acc2 = __builtin_amdgcn_wmma_f32_16x16x32_f16(false, au.v, false, b2.v, (short)0, acc2, false, false);
        acc3 = __builtin_amdgcn_wmma_f32_16x16x32_f16(false, au.v, false, b3.v, (short)0, acc3, false, false);
    }
    // C/D layout: VGPR r -> row r (lanes 0-15) / r+8 (lanes 16-31), col = lane&15
    int mr = m0 + ((lane >> 4) << 3), cc = lane & 15;
    for (int r = 0; r < 8; r++) {
        float* cp = C + (size_t)(mr + r) * Ncp + n0 + cc;
        cp[0] = acc0[r]; cp[16] = acc1[r]; cp[32] = acc2[r]; cp[48] = acc3[r];
    }
}

// ---------------- bias + SELU over padded matrix ----------------
__global__ void k_selu_bias(float* z, const float* __restrict__ b,
                            int rows, int Nc, int Ncp) {
    int t = blockIdx.x * blockDim.x + threadIdx.x;
    if (t >= rows * Ncp) return;
    int c = t % Ncp;
    float v = z[t] + (c < Nc ? b[c] : 0.f);
    z[t] = selu_f(v);
}

// ---------------- LFE scatter-add into nodes ----------------
__global__ void k_scatter_lfe(const float* __restrict__ z, const int* __restrict__ ei,
                              float* __restrict__ l) {
    int t = blockIdx.x * blockDim.x + threadIdx.x;
    if (t >= EE * 16) return;
    int e = t >> 4, c = t & 15;
    atomicAdd(&l[ei[EE + e] * 16 + c], z[e * 64 + c]);
}

// ---------------- h0 = [g | l | x] ----------------
__global__ void k_build_h0(const float* __restrict__ g, const float* __restrict__ l,
                           const float* __restrict__ x, float* __restrict__ h) {
    int t = blockIdx.x * blockDim.x + threadIdx.x;
    if (t >= NN * 42) return;
    int n = t / 42, c = t - n * 42;
    h[t] = c < 16 ? g[c] : (c < 32 ? l[n * 16 + c - 16] : x[n * 10 + c - 32]);
}

// ---------------- GAT attention projections ----------------
__global__ void k_att(const float* __restrict__ xp, const float* __restrict__ asrc,
                      const float* __restrict__ adst, float* __restrict__ as_,
                      float* __restrict__ ad_, int H, int fo, int Ncp) {
    int t = blockIdx.x * blockDim.x + threadIdx.x;
    if (t >= NN * H) return;
    int n = t / H, hd = t - n * H;
    const float* xr = xp + (size_t)n * Ncp + hd * fo;
    const float* s0 = asrc + hd * fo;
    const float* d0 = adst + hd * fo;
    float s = 0.f, d = 0.f;
    for (int c = 0; c < fo; c++) { float v = xr[c]; s += v * s0[c]; d += v * d0[c]; }
    as_[t] = s; ad_[t] = d;
}

__global__ void k_edge_max(const int* __restrict__ ei, const float* __restrict__ as_,
                           const float* __restrict__ ad_, float* __restrict__ ebuf,
                           unsigned* __restrict__ mx, int H) {
    int t = blockIdx.x * blockDim.x + threadIdx.x;
    if (t >= ET * H) return;
    int e = t / H, hd = t - e * H;
    int s, d; edge_sd(ei, e, s, d);
    float v = as_[s * H + hd] + ad_[d * H + hd];
    v = v > 0.f ? v : 0.2f * v;                  // leaky_relu(0.2)
    ebuf[t] = v;
    atomicMax(&mx[d * H + hd], enc_f(v));
}

__global__ void k_edge_alpha(const int* __restrict__ ei, const float* __restrict__ ebuf,
                             const unsigned* __restrict__ mx, float* __restrict__ abuf,
                             float* __restrict__ den, int H) {
    int t = blockIdx.x * blockDim.x + threadIdx.x;
    if (t >= ET * H) return;
    int e = t / H, hd = t - e * H;
    int s, d; edge_sd(ei, e, s, d);
    float a = __expf(ebuf[t] - dec_f(mx[d * H + hd]));
    abuf[t] = a;
    atomicAdd(&den[d * H + hd], a);
}

__global__ void k_edge_aggr(const int* __restrict__ ei, const float* __restrict__ xp,
                            const float* __restrict__ abuf, const float* __restrict__ den,
                            float* __restrict__ acc, int H, int fo, int HC, int Ncp) {
    int t = blockIdx.x * blockDim.x + threadIdx.x;
    int hc4 = HC >> 2;
    if (t >= ET * hc4) return;
    int e = t / hc4, q = t - e * hc4, c0 = q << 2;
    int s, d; edge_sd(ei, e, s, d);
    const float4 xv = *(const float4*)(xp + (size_t)s * Ncp + c0);
    float vx[4] = { xv.x, xv.y, xv.z, xv.w };
    for (int j = 0; j < 4; j++) {
        int c = c0 + j, hd = c / fo;
        float w = abuf[e * H + hd] / den[d * H + hd];
        atomicAdd(&acc[(size_t)d * HC + c], vx[j] * w);
    }
}

__global__ void k_node_fin(const float* __restrict__ acc, const float* __restrict__ bias,
                           const float* __restrict__ x, float* __restrict__ hout,
                           int H, int fo, int HC, int dofix) {
    int t = blockIdx.x * blockDim.x + threadIdx.x;
    if (t >= NN * fo) return;
    int n = t / fo, c = t - n * fo;
    float s = 0.f;
    for (int hd = 0; hd < H; hd++) s += acc[(size_t)n * HC + hd * fo + c];
    float v = selu_f(s / (float)H + bias[c]);
    if (dofix) {
        if (c == 0) { float x0 = x[n * 10];     v = (x0 == 0.f) ? 0.f : ((x0 == 1.f) ? 1.f : v); }
        if (c == 1) { float x1 = x[n * 10 + 1]; v = (x1 == 1.f) ? 1.f : ((x1 == 0.f) ? 0.f : v); }
    }
    hout[t] = v;
}

__global__ void k_final(const float* __restrict__ h, const float* __restrict__ x,
                        float* __restrict__ out) {
    int t = blockIdx.x * blockDim.x + threadIdx.x;
    if (t >= NN) return;
    float x0 = x[t * 10], x1 = x[t * 10 + 1];
    float o0 = x0 + h[2 * t], o1 = x1 + h[2 * t + 1];
    o0 = (x0 == 0.f) ? 0.f : ((x0 == 1.f) ? 1.f : o0);
    o1 = (x1 == 1.f) ? 1.f : ((x1 == 0.f) ? 0.f : o1);
    out[2 * t] = o0;
    out[2 * t + 1] = o1;
}

extern "C" void kernel_launch(void* const* d_in, const int* in_sizes, int n_in,
                              void* d_out, int out_size, void* d_ws, size_t ws_size,
                              hipStream_t stream) {
    const float* x  = (const float*)d_in[0];
    const int*   ei = (const int*)d_in[1];
    const float* cf = (const float*)d_in[2];
    const float *gw[4], *gbv[4];
    for (int i = 0; i < 4; i++) { gw[i] = (const float*)d_in[3 + 2 * i]; gbv[i] = (const float*)d_in[4 + 2 * i]; }
    const float *lw[3], *lb[3];
    for (int i = 0; i < 3; i++) { lw[i] = (const float*)d_in[11 + 2 * i]; lb[i] = (const float*)d_in[12 + 2 * i]; }
    const float *gW[5], *gas[5], *gad[5], *gbias[5];
    for (int i = 0; i < 5; i++) {
        gW[i]    = (const float*)d_in[17 + 4 * i];
        gas[i]   = (const float*)d_in[18 + 4 * i];
        gad[i]   = (const float*)d_in[19 + 4 * i];
        gbias[i] = (const float*)d_in[20 + 4 * i];
    }

    uint8_t* wsp = (uint8_t*)d_ws;
    auto take = [&](size_t bytes) -> void* {
        void* p = (void*)wsp;
        wsp += (bytes + 255) & ~(size_t)255;
        return p;
    };
    float*     xp   = (float*)take((size_t)NN * 2048 * 4);
    float*     accb = (float*)take((size_t)NN * 2048 * 4);
    float*     hA   = (float*)take((size_t)NN * 256 * 4);
    float*     hB   = (float*)take((size_t)NN * 256 * 4);
    _Float16*  h16  = (_Float16*)take((size_t)NN * 256 * 2);
    _Float16*  B16  = (_Float16*)take((size_t)256 * 2048 * 2);
    float*     as_  = (float*)take((size_t)NN * 16 * 4);
    float*     ad_  = (float*)take((size_t)NN * 16 * 4);
    unsigned*  mx   = (unsigned*)take((size_t)NN * 16 * 4);
    float*     den  = (float*)take((size_t)NN * 16 * 4);
    float*     lbuf = (float*)take((size_t)NN * 16 * 4);
    float*     ebuf = (float*)take((size_t)ET * 16 * 4);
    float*     abuf = (float*)take((size_t)ET * 16 * 4);
    float*     cb1  = (float*)take((size_t)32 * 4096 * 4);
    float*     cb2  = (float*)take((size_t)64 * 4096 * 4);
    float*     cb3  = (float*)take((size_t)64 * 4096 * 4);
    float*     cb4  = (float*)take((size_t)16 * 4096 * 4);
    float*     gvec = (float*)take(16 * 4);

#define GL(kern, n, ...) kern<<<((n) + 255) / 256, 256, 0, stream>>>(__VA_ARGS__)
    auto gemmLaunch = [&](const _Float16* A, const _Float16* Bp, float* Cc,
                          int M, int Kp, int Ncp) {
        int blocks = ((M / 16 + 7) / 8) * (Ncp >> 6);
        k_gemm_wmma<<<blocks, 256, 0, stream>>>(A, Bp, Cc, M, Kp, Ncp);
    };

    // ---- GFE (tiny)
    GL(k_conv3x3_selu, 32 * 4096, cf, gw[0], gbv[0], cb1, 4, 32);
    GL(k_conv3x3_selu, 64 * 4096, cb1, gw[1], gbv[1], cb2, 32, 64);
    GL(k_conv3x3_selu, 64 * 4096, cb2, gw[2], gbv[2], cb3, 64, 64);
    GL(k_conv3x3_selu, 16 * 4096, cb3, gw[3], gbv[3], cb4, 64, 16);
    k_avgpool16<<<16, 256, 0, stream>>>(cb4, gvec);

    // ---- LFE edge MLP via WMMA (E rows, E % 16 == 0)
    GL(k_build_m16, EE * 32, x, ei, h16);
    GL(k_repackB, (32 >> 5) * (64 >> 4) * 512, lw[0], B16, 30, 64, 32, 64);
    gemmLaunch(h16, B16, xp, EE, 32, 64);
    GL(k_selu_bias, EE * 64, xp, lb[0], EE, 64, 64);
    GL(k_f2h_pad, EE * 64, xp, h16, EE, 64, 64);
    GL(k_repackB, (64 >> 5) * (64 >> 4) * 512, lw[1], B16, 64, 64, 64, 64);
    gemmLaunch(h16, B16, accb, EE, 64, 64);
    GL(k_selu_bias, EE * 64, accb, lb[1], EE, 64, 64);
    GL(k_f2h_pad, EE * 64, accb, h16, EE, 64, 64);
    GL(k_repackB, (64 >> 5) * (64 >> 4) * 512, lw[2], B16, 64, 16, 64, 64);
    gemmLaunch(h16, B16, xp, EE, 64, 64);
    GL(k_selu_bias, EE * 64, xp, lb[2], EE, 16, 64);
    GL(k_zero, NN * 16, lbuf, NN * 16);
    GL(k_scatter_lfe, EE * 16, xp, ei, lbuf);
    GL(k_build_h0, NN * 42, gvec, lbuf, x, hA);

    // ---- 5 GAT layers
    const int fiA[5] = { 42, 64, 128, 256, 128 };
    const int foA[5] = { 64, 128, 256, 128, 2 };
    const int HA[5]  = { 8, 16, 8, 8, 16 };
    float* hin = hA; float* hout = hB;
    for (int i = 0; i < 5; i++) {
        int fi = fiA[i], fo = foA[i], H = HA[i];
        int HC = H * fo;
        int Kp = ((fi + 31) / 32) * 32;
        int Ncp = ((HC + 63) / 64) * 64;
        GL(k_repackB, (Kp >> 5) * (Ncp >> 4) * 512, gW[i], B16, fi, HC, Kp, Ncp);
        GL(k_f2h_pad, NN * Kp, hin, h16, NN, fi, Kp);
        gemmLaunch(h16, B16, xp, NN, Kp, Ncp);
        GL(k_att, NN * H, xp, gas[i], gad[i], as_, ad_, H, fo, Ncp);
        GL(k_zero, NN * H, (float*)mx, NN * H);     // enc==0 -> below everything
        GL(k_zero, NN * H, den, NN * H);
        GL(k_edge_max, ET * H, ei, as_, ad_, ebuf, mx, H);
        GL(k_edge_alpha, ET * H, ei, ebuf, mx, abuf, den, H);
        GL(k_zero, NN * HC, accb, NN * HC);
        GL(k_edge_aggr, ET * (HC / 4), ei, xp, abuf, den, accb, H, fo, HC, Ncp);
        GL(k_node_fin, NN * fo, accb, gbias[i], x, hout, H, fo, HC, (i < 4) ? 1 : 0);
        float* tmp = hin; hin = hout; hout = tmp;
    }
    GL(k_final, NN, hin, x, (float*)d_out);
#undef GL
}
